// QGSN_sparse_58737972740097
// MI455X (gfx1250) — matrix operand order repaired and checked by reading, compiled, not run on hardware
//
#include <hip/hip_runtime.h>

// ---------------------------------------------------------------------------
// GNN message-passing layer for MI455X (gfx1250, wave32, WMMA bf16)
//   edge MLP  : [E,192] x [192,192] -> ReLU -> x [192,64], scale, atomic scatter
//   node MLP  : [N,128] x [128,192] -> ReLU -> x [192,128] -> out
// Weights converted once per block to bf16 in LDS, transposed to [n][k] so a
// B-fragment is two runs of 8 contiguous K values per lane. Row tiles of 16 per wave.
// All LDS row strides padded +8 bf16 (16B) so 16 lanes hit 16 distinct banks
// (64-bank LDS) while keeping ds_load_b128 alignment.
// ---------------------------------------------------------------------------

typedef __attribute__((ext_vector_type(16))) __bf16 v16bf;
typedef __attribute__((ext_vector_type(8)))  float  v8f;

constexpr int DIN   = 64;
constexpr int DID   = 32;
constexpr int DH    = 192;
constexpr int DMSG  = 64;
constexpr int DUP   = 128;
constexpr int MSGIN = 2 * (DIN + DID);   // 192
constexpr int UPDIN = DIN + DMSG;        // 128

constexpr int PAD    = 8;                // +16 bytes per row: bank spread, 16B aligned
constexpr int S192   = DH + PAD;         // 200 bf16 = 100 dw -> stride 36 mod 64
constexpr int S128   = UPDIN + PAD;      // 136 bf16 = 68 dw  -> stride 4 mod 64

constexpr int WPB = 4;                   // waves per block
constexpr int TPB = WPB * 32;

struct FragU { uint4 lo, hi; };          // 32 bytes == v16bf

__device__ __forceinline__ v16bf frag_cast(FragU f) {
  return __builtin_bit_cast(v16bf, f);
}

__device__ __forceinline__ unsigned short f2bf(float f) {
  union { float f; unsigned u; } c; c.f = f;
  unsigned u = c.u;
  u += 0x7FFFu + ((u >> 16) & 1u);       // round-to-nearest-even
  return (unsigned short)(u >> 16);
}

__device__ __forceinline__ v8f zero8() {
  v8f z = {0.f, 0.f, 0.f, 0.f, 0.f, 0.f, 0.f, 0.f};
  return z;
}

// A-fragment (16x32 bf16, M x K) from LDS stage[16][rowstride].
// ISA layout: lanes 0-15 & 16-31 both hold M = lane&15; lane>>4 selects the
// K sub-interleave {0..7,16..23} vs {8..15,24..31}.  16B-aligned ds_load_b128.
__device__ __forceinline__ v16bf load_a_frag(const unsigned short* stage,
                                             int rowstride, int kb, int lane) {
  const int m  = lane & 15;
  const int hi = lane >> 4;
  const unsigned short* p = stage + m * rowstride + kb * 32 + hi * 8;
  FragU f;
  f.lo = *(const uint4*)(p);        // K = 32*kb + 8*hi + {0..7}
  f.hi = *(const uint4*)(p + 16);   // K = 32*kb + 8*hi + {16..23}
  return frag_cast(f);
}

// B-fragment (32x16 bf16, K x N) from LDS wT[n][Kstride] (K contiguous).
// Lane holds column n = nt*16 + (lane&15); lanes>=16 hold K base +8.
__device__ __forceinline__ v16bf load_b_frag(const unsigned short* wT,
                                             int kstride, int nt, int kb, int lane) {
  const int n     = nt * 16 + (lane & 15);
  const int kbase = kb * 32 + (lane >> 4) * 8;
  const unsigned short* p = wT + n * kstride + kbase;
  FragU f;
  f.lo = *(const uint4*)(p);        // k = kbase + 0..7
  f.hi = *(const uint4*)(p + 16);   // k = kbase + 16..23
  return frag_cast(f);
}

__device__ __forceinline__ v8f wmma_bf16(v16bf a, v16bf b, v8f c) {
  return __builtin_amdgcn_wmma_f32_16x16x32_bf16(false, a, false, b,
                                                 (short)0, c, false, false);
}

// ---------------------------------------------------------------------------
// Edge kernel: per-wave 16-edge tile, gather -> GEMM1 -> ReLU -> GEMM2 ->
// scale by edge_centrality -> atomicAdd scatter into agg[N,64].
// ---------------------------------------------------------------------------
__global__ void __launch_bounds__(TPB)
gnn_edge_kernel(const float* __restrict__ x,
                const long long* __restrict__ edge_index,  // [2,E] int64
                const float* __restrict__ edge_c,
                const float* __restrict__ ident,
                const float* __restrict__ W1, const float* __restrict__ b1,
                const float* __restrict__ W2, const float* __restrict__ b2,
                float* __restrict__ agg,
                int nEdges) {
  extern __shared__ unsigned short smem[];
  unsigned short* w1T       = smem;                    // [192][S192] bf16
  unsigned short* w2T       = w1T + DH * S192;         // [64][S192]  bf16
  unsigned short* stage_all = w2T + DMSG * S192;       // WPB * 16*S192 bf16

  // Convert weights to bf16, transposed to [n][k] (one-time per block).
  for (int i = threadIdx.x; i < DH * MSGIN; i += blockDim.x) {
    const int n = i / MSGIN, k = i % MSGIN;
    w1T[n * S192 + k] = f2bf(W1[k * DH + n]);
  }
  for (int i = threadIdx.x; i < DMSG * DH; i += blockDim.x) {
    const int n = i / DH, k = i % DH;
    w2T[n * S192 + k] = f2bf(W2[k * DMSG + n]);
  }
  __syncthreads();

  const int lane = threadIdx.x & 31;
  const int wid  = threadIdx.x >> 5;
  const int hi   = lane >> 4;
  const int nn   = lane & 15;
  unsigned short* stage = stage_all + wid * 16 * S192;

  const int nTiles     = (nEdges + 15) / 16;
  const int waveId     = blockIdx.x * WPB + wid;
  const int waveStride = gridDim.x * WPB;

  for (int tile = waveId; tile < nTiles; tile += waveStride) {
    const int ebase = tile * 16;

    // ---- gather [x_i, x_j, id_i, id_j] -> bf16 stage[16][S192] ----
    #pragma unroll 1
    for (int s = 0; s < 16; ++s) {
      int e = ebase + s;
      if (e >= nEdges) e = nEdges - 1;
      const int src = (int)edge_index[e];             // j = edge_index[0]
      const int tgt = (int)edge_index[nEdges + e];    // i = edge_index[1]
      for (int f = lane; f < MSGIN; f += 32) {
        float v;
        if      (f < DIN)            v = x[(size_t)tgt * DIN + f];
        else if (f < 2 * DIN)        v = x[(size_t)src * DIN + (f - DIN)];
        else if (f < 2 * DIN + DID)  v = ident[(size_t)tgt * DID + (f - 2 * DIN)];
        else                         v = ident[(size_t)src * DID + (f - 2 * DIN - DID)];
        stage[s * S192 + f] = f2bf(v);
      }
    }

    // ---- GEMM1: [16,192] x [192,192] ----
    v8f acc1[DH / 16];
    #pragma unroll
    for (int nt = 0; nt < DH / 16; ++nt) acc1[nt] = zero8();
    #pragma unroll
    for (int kb = 0; kb < MSGIN / 32; ++kb) {
      const v16bf a = load_a_frag(stage, S192, kb, lane);
      #pragma unroll
      for (int nt = 0; nt < DH / 16; ++nt) {
        const v16bf b = load_b_frag(w1T, S192, nt, kb, lane);
        acc1[nt] = wmma_bf16(a, b, acc1[nt]);
      }
    }

    // ---- bias + ReLU, restage hidden [16][S192] as bf16 ----
    #pragma unroll
    for (int nt = 0; nt < DH / 16; ++nt) {
      const float bias = b1[nt * 16 + nn];
      #pragma unroll
      for (int r = 0; r < 8; ++r) {
        float h = acc1[nt][r] + bias;       // C layout: m = r + 8*hi, n = nn
        h = fmaxf(h, 0.0f);
        stage[(r + 8 * hi) * S192 + nt * 16 + nn] = f2bf(h);
      }
    }

    // ---- GEMM2: [16,192] x [192,64] ----
    v8f acc2[DMSG / 16];
    #pragma unroll
    for (int nt = 0; nt < DMSG / 16; ++nt) acc2[nt] = zero8();
    #pragma unroll
    for (int kb = 0; kb < DH / 32; ++kb) {
      const v16bf a = load_a_frag(stage, S192, kb, lane);
      #pragma unroll
      for (int nt = 0; nt < DMSG / 16; ++nt) {
        const v16bf b = load_b_frag(w2T, S192, nt, kb, lane);
        acc2[nt] = wmma_bf16(a, b, acc2[nt]);
      }
    }

    // ---- bias + edge_centrality scale + atomic segment-sum ----
    #pragma unroll
    for (int r = 0; r < 8; ++r) {
      const int m = r + 8 * hi;
      const int e = ebase + m;
      if (e < nEdges) {
        const int   tgt = (int)edge_index[nEdges + e];
        const float ec  = edge_c[e];
        #pragma unroll
        for (int nt = 0; nt < DMSG / 16; ++nt) {
          const float v = (acc2[nt][r] + b2[nt * 16 + nn]) * ec;
          atomicAdd(&agg[(size_t)tgt * DMSG + nt * 16 + nn], v);
        }
      }
    }
  }
}

// ---------------------------------------------------------------------------
// Node kernel: per-wave 16-node tile, [x*c, agg] -> GEMM1 -> ReLU -> GEMM2
// ---------------------------------------------------------------------------
__global__ void __launch_bounds__(TPB)
gnn_node_kernel(const float* __restrict__ x,
                const float* __restrict__ node_c,
                const float* __restrict__ agg,
                const float* __restrict__ W1, const float* __restrict__ b1,
                const float* __restrict__ W2, const float* __restrict__ b2,
                float* __restrict__ out,
                int nNodes) {
  extern __shared__ unsigned short smem[];
  unsigned short* w1T       = smem;                    // [192][S128] bf16
  unsigned short* w2T       = w1T + DH * S128;         // [128][S192] bf16
  unsigned short* stage_all = w2T + DUP * S192;        // WPB * 16*S192 bf16

  for (int i = threadIdx.x; i < DH * UPDIN; i += blockDim.x) {
    const int n = i / UPDIN, k = i % UPDIN;
    w1T[n * S128 + k] = f2bf(W1[k * DH + n]);
  }
  for (int i = threadIdx.x; i < DUP * DH; i += blockDim.x) {
    const int n = i / DH, k = i % DH;
    w2T[n * S192 + k] = f2bf(W2[k * DUP + n]);
  }
  __syncthreads();

  const int lane = threadIdx.x & 31;
  const int wid  = threadIdx.x >> 5;
  const int hi   = lane >> 4;
  const int nn   = lane & 15;
  unsigned short* stage = stage_all + wid * 16 * S192;

  const int nTiles     = (nNodes + 15) / 16;
  const int waveId     = blockIdx.x * WPB + wid;
  const int waveStride = gridDim.x * WPB;

  for (int tile = waveId; tile < nTiles; tile += waveStride) {
    const int nbase = tile * 16;

    // ---- stage upd_in = [x * centrality, agg] -> bf16 [16][S128] ----
    #pragma unroll 1
    for (int s = 0; s < 16; ++s) {
      int nd = nbase + s;
      if (nd >= nNodes) nd = nNodes - 1;
      const float nc = node_c[nd];
      for (int f = lane; f < UPDIN; f += 32) {
        const float v = (f < DIN) ? x[(size_t)nd * DIN + f] * nc
                                  : agg[(size_t)nd * DMSG + (f - DIN)];
        stage[s * S128 + f] = f2bf(v);
      }
    }

    // ---- GEMM1: [16,128] x [128,192] ----
    v8f acc1[DH / 16];
    #pragma unroll
    for (int nt = 0; nt < DH / 16; ++nt) acc1[nt] = zero8();
    #pragma unroll
    for (int kb = 0; kb < UPDIN / 32; ++kb) {
      const v16bf a = load_a_frag(stage, S128, kb, lane);
      #pragma unroll
      for (int nt = 0; nt < DH / 16; ++nt) {
        const v16bf b = load_b_frag(w1T, S128, nt, kb, lane);
        acc1[nt] = wmma_bf16(a, b, acc1[nt]);
      }
    }

    // ---- bias + ReLU, restage hidden [16][S192] ----
    #pragma unroll
    for (int nt = 0; nt < DH / 16; ++nt) {
      const float bias = b1[nt * 16 + nn];
      #pragma unroll
      for (int r = 0; r < 8; ++r) {
        float h = acc1[nt][r] + bias;
        h = fmaxf(h, 0.0f);
        stage[(r + 8 * hi) * S192 + nt * 16 + nn] = f2bf(h);
      }
    }

    // ---- GEMM2: [16,192] x [192,128] ----
    v8f acc2[DUP / 16];
    #pragma unroll
    for (int nt = 0; nt < DUP / 16; ++nt) acc2[nt] = zero8();
    #pragma unroll
    for (int kb = 0; kb < DH / 32; ++kb) {
      const v16bf a = load_a_frag(stage, S192, kb, lane);
      #pragma unroll
      for (int nt = 0; nt < DUP / 16; ++nt) {
        const v16bf b = load_b_frag(w2T, S192, nt, kb, lane);
        acc2[nt] = wmma_bf16(a, b, acc2[nt]);
      }
    }

    // ---- bias + store out [16,128] f32 ----
    #pragma unroll
    for (int nt = 0; nt < DUP / 16; ++nt) {
      const float bias = b2[nt * 16 + nn];
      #pragma unroll
      for (int r = 0; r < 8; ++r) {
        const int m  = r + 8 * hi;
        const int nd = nbase + m;
        if (nd < nNodes)
          out[(size_t)nd * DUP + nt * 16 + nn] = acc2[nt][r] + bias;
      }
    }
  }
}

// ---------------------------------------------------------------------------
extern "C" void kernel_launch(void* const* d_in, const int* in_sizes, int n_in,
                              void* d_out, int out_size, void* d_ws, size_t ws_size,
                              hipStream_t stream) {
  const float*     x      = (const float*)d_in[0];
  const long long* ei     = (const long long*)d_in[1];   // int64 [2,E]
  const float*     node_c = (const float*)d_in[2];
  const float*     edge_c = (const float*)d_in[3];
  const float*     ident  = (const float*)d_in[4];
  // d_in[5] = degrees (unused by reference)
  const float* mW1 = (const float*)d_in[6];
  const float* mb1 = (const float*)d_in[7];
  const float* mW2 = (const float*)d_in[8];
  const float* mb2 = (const float*)d_in[9];
  const float* uW1 = (const float*)d_in[10];
  const float* ub1 = (const float*)d_in[11];
  const float* uW2 = (const float*)d_in[12];
  const float* ub2 = (const float*)d_in[13];
  float* out = (float*)d_out;

  const int nNodes = in_sizes[2];   // node_centrality length
  const int nEdges = in_sizes[3];   // edge_centrality length

  float* agg = (float*)d_ws;        // [nNodes, 64] f32 segment-sum accumulator
  hipMemsetAsync(agg, 0, (size_t)nNodes * DMSG * sizeof(float), stream);

  const size_t ldsEdge =
      (size_t)(DH * S192 + DMSG * S192 + WPB * 16 * S192) * sizeof(unsigned short);
  const size_t ldsNode =
      (size_t)(DH * S128 + DUP * S192 + WPB * 16 * S192) * sizeof(unsigned short);

  hipFuncSetAttribute((const void*)gnn_edge_kernel,
                      hipFuncAttributeMaxDynamicSharedMemorySize, (int)ldsEdge);
  hipFuncSetAttribute((const void*)gnn_node_kernel,
                      hipFuncAttributeMaxDynamicSharedMemorySize, (int)ldsNode);

  const int edgeTiles  = (nEdges + 15) / 16;
  int edgeBlocks = (edgeTiles + WPB - 1) / WPB;
  if (edgeBlocks > 640) edgeBlocks = 640;
  gnn_edge_kernel<<<edgeBlocks, TPB, ldsEdge, stream>>>(
      x, ei, edge_c, ident, mW1, mb1, mW2, mb2, agg, nEdges);

  const int nodeTiles  = (nNodes + 15) / 16;
  int nodeBlocks = (nodeTiles + WPB - 1) / WPB;
  if (nodeBlocks > 640) nodeBlocks = 640;
  gnn_node_kernel<<<nodeBlocks, TPB, ldsNode, stream>>>(
      x, node_c, agg, uW1, ub1, uW2, ub2, out, nNodes);
}
